// CAPMemory_online_44607530336553
// MI455X (gfx1250) — matrix operand order repaired and checked by reading, not compiled, not run on hardware
//
#include <hip/hip_runtime.h>
#include <math.h>

// ---------------- problem dims (match reference) ----------------
#define BB      256      // batch
#define DD      2048     // feature dim (K)
#define NCAMS   6
#define PCAM    8192
#define MTOT    49152    // NCAMS*PCAM
#define NLABELS 4096
#define AASSOC  6
#define BG_KNN  50
#define POS_K   3
#define TOP_EXTRA 30
#define TOPN    56       // BG_KNN + AASSOC  (covers both exclusion needs)
#define INV_BETA 20.0f   // 1/0.05
#define NBAND   16       // 256 rows = 16 bands of 16

// ---------------- CDNA5 vector types ----------------
typedef __attribute__((ext_vector_type(16))) __bf16        v16bf;
typedef __attribute__((ext_vector_type(8)))  float         v8f;
typedef __attribute__((ext_vector_type(8)))  unsigned short ush8;
typedef __attribute__((ext_vector_type(4)))  float         f32x4;

union ABv { v16bf v; ush8 h[2]; unsigned short u[16]; unsigned int w[8]; };

// round-to-nearest bf16 pair, packed with one v_perm_b32
__device__ __forceinline__ unsigned int pack_bf16_rn(float hi, float lo) {
  unsigned int a = __float_as_uint(lo) + 0x8000u;   // low 16 of result
  unsigned int b = __float_as_uint(hi) + 0x8000u;   // high 16 of result
  // dst bytes {3,2,1,0} = {S0.b3, S0.b2, S1.b3, S1.b2}  (S0=b, S1=a)
  return __builtin_amdgcn_perm(b, a, 0x07060302u);
}

__device__ __forceinline__ unsigned short f32_to_bf16(float f) {
  unsigned int r = __float_as_uint(f) + 0x8000u;    // round-to-nearest
  return (unsigned short)(r >> 16);
}

// total-order key: value (monotonic bits) major, lower index wins ties
__device__ __forceinline__ unsigned long long key_of(float v, int idx) {
  unsigned int b = __float_as_uint(v);
  unsigned int u = (b & 0x80000000u) ? ~b : (b | 0x80000000u);
  return ((unsigned long long)u << 16) |
         (unsigned long long)(0xFFFFu - (unsigned int)idx);
}
__device__ __forceinline__ int key_idx(unsigned long long k) {
  return 0xFFFF - (int)(k & 0xFFFFull);
}
__device__ __forceinline__ float key_val(unsigned long long k) {
  unsigned int u = (unsigned int)(k >> 16);
  unsigned int b = (u & 0x80000000u) ? (u & 0x7FFFFFFFu) : ~u;
  return __uint_as_float(b);
}

__device__ __forceinline__ void insert8(unsigned long long* tk, unsigned long long k) {
  if (k <= tk[7]) return;
  tk[7] = k;
#pragma unroll
  for (int q = 7; q > 0; --q) {
    if (tk[q] > tk[q - 1]) {
      unsigned long long t = tk[q - 1]; tk[q - 1] = tk[q]; tk[q] = t;
    }
  }
}

// ---------------- kernel 0: features f32 -> bf16 (packed, 2 elems/thread) ----
__global__ __launch_bounds__(256) void k_feat_bf16(const float* __restrict__ src,
                                                   unsigned int* __restrict__ dst,
                                                   int npairs) {
  int i = blockIdx.x * 256 + threadIdx.x;
  if (i < npairs) {
    float lo = src[2 * i], hi = src[2 * i + 1];
    dst[i] = pack_bf16_rn(hi, lo);
  }
}

// ---------------- kernel 1: sims = features @ bank^T via WMMA bf16 ----------
// grid: MTOT/128 blocks; block: 256 = 8 waves.
// Each wave owns a 16-column strip and accumulates ALL 256 rows (16 bands),
// so every bank element is read from HBM exactly once and every converted
// B-fragment is reused by 16 WMMAs from registers.
__global__ __launch_bounds__(256, 1) void k_wmma_gemm(const unsigned short* __restrict__ fb,
                                                      const float* __restrict__ bank,
                                                      float* __restrict__ sims) {
  const int lane = threadIdx.x & 31;
  const int wv   = threadIdx.x >> 5;
  const int hh   = lane >> 4;          // lane half (K-half selector)
  const int l16  = lane & 15;
  const int colBase = (blockIdx.x << 7) + (wv << 4);

  const float*          brow  = bank + (size_t)(colBase + l16) * DD;
  const unsigned short* abase = fb + (size_t)l16 * DD + hh * 16;   // + kb later

  v8f acc[NBAND];
#pragma unroll
  for (int b = 0; b < NBAND; ++b) acc[b] = (v8f){};

  for (int kb = 0; kb < DD; kb += 32) {
    // ---- B fragment: 32x16, 16 contiguous f32 per lane -> packed bf16 ----
    const float* bp = brow + kb + hh * 16;
    __builtin_prefetch(bp + 128, 0, 0);
    ABv Bm;
#pragma unroll
    for (int q = 0; q < 4; ++q) {
      f32x4 f = *(const f32x4*)(bp + q * 4);
      Bm.w[2 * q]     = pack_bf16_rn(f.y, f.x);
      Bm.w[2 * q + 1] = pack_bf16_rn(f.w, f.z);
    }
    // ---- 16 row bands: load A fragment (WGP$/L2 hot) and accumulate ----
    const unsigned short* ak = abase + kb;
#pragma unroll
    for (int band = 0; band < NBAND; ++band) {
      ABv A;
      A.h[0] = *(const ush8*)(ak + band * (16 * DD));
      A.h[1] = *(const ush8*)(ak + band * (16 * DD) + 8);
      acc[band] = __builtin_amdgcn_wmma_f32_16x16x32_bf16(
          /*neg_a=*/false, A.v, /*neg_b=*/false, Bm.v,
          /*c_mod=*/(short)0, acc[band], /*reuse_a=*/false, /*reuse_b=*/false);
    }
  }
  // C/D layout: VGPR r -> M = r + 8*hh (within band), N = lane&15
#pragma unroll
  for (int band = 0; band < NBAND; ++band) {
#pragma unroll
    for (int r = 0; r < 8; ++r) {
      int mm = band * 16 + r + hh * 8;
      sims[(size_t)mm * MTOT + colBase + l16] = acc[band][r];
    }
  }
}

// ---------------- kernel 2: per-row losses ----------------
__global__ __launch_bounds__(256) void k_row(const float* __restrict__ sims,
                                             const int* __restrict__ proxy,
                                             const int* __restrict__ cams,
                                             const int* __restrict__ plabel,
                                             const int* __restrict__ assoc_tab,
                                             float* __restrict__ ce_out,
                                             float* __restrict__ as_out,
                                             float* __restrict__ on_out) {
  __shared__ unsigned long long s_k[256];
  __shared__ unsigned long long s_cand[256 * 8];
  __shared__ unsigned long long s_camKey[NCAMS];
  __shared__ float s_f[256];
  __shared__ float s_topVal[TOPN];
  __shared__ int   s_topIdx[TOPN];

  const int i   = blockIdx.x;
  const int tid = threadIdx.x;
  const float* row = sims + (size_t)i * MTOT;

  // ---- pass 1: per-thread top-8 + per-cam argmax ----
  unsigned long long tk[8] = {0, 0, 0, 0, 0, 0, 0, 0};
  unsigned long long camk[NCAMS] = {0, 0, 0, 0, 0, 0};
  for (int j = tid; j < MTOT; j += 256) {
    unsigned long long k = key_of(row[j], j);
    insert8(tk, k);
    int c = j >> 13;                   // PCAM = 2^13
    if (k > camk[c]) camk[c] = k;
  }
#pragma unroll
  for (int q = 0; q < 8; ++q) s_cand[(tid << 3) + q] = tk[q];

  // ---- per-cam argmax reductions ----
  for (int c = 0; c < NCAMS; ++c) {
    s_k[tid] = camk[c];
    __syncthreads();
    for (int s = 128; s > 0; s >>= 1) {
      if (tid < s) { if (s_k[tid + s] > s_k[tid]) s_k[tid] = s_k[tid + s]; }
      __syncthreads();
    }
    if (tid == 0) s_camKey[c] = s_k[0];
    __syncthreads();
  }

  // ---- extract global top-56 (value,index) ----
  for (int t = 0; t < TOPN; ++t) {
    unsigned long long best = 0;
#pragma unroll
    for (int q = 0; q < 8; ++q) {
      unsigned long long k = s_cand[(tid << 3) + q];
      if (k > best) best = k;
    }
    s_k[tid] = best;
    __syncthreads();
    for (int s = 128; s > 0; s >>= 1) {
      if (tid < s) { if (s_k[tid + s] > s_k[tid]) s_k[tid] = s_k[tid + s]; }
      __syncthreads();
    }
    unsigned long long K = s_k[0];
    __syncthreads();
#pragma unroll
    for (int q = 0; q < 8; ++q) {
      if (s_cand[(tid << 3) + q] == K) s_cand[(tid << 3) + q] = 0;
    }
    if (tid == 0) { s_topIdx[t] = key_idx(K); s_topVal[t] = key_val(K); }
    __syncthreads();
  }

  // ---- part 1: cam-block logsumexp ----
  const int cam  = cams[i];
  const int base = cam * PCAM;
  const float m1 = key_val(s_camKey[cam]);   // block max
  float part = 0.0f;
  for (int j = base + tid; j < base + PCAM; j += 256)
    part += __expf((row[j] - m1) * INV_BETA);
  s_f[tid] = part;
  __syncthreads();
  for (int s = 128; s > 0; s >>= 1) {
    if (tid < s) s_f[tid] += s_f[tid + s];
    __syncthreads();
  }

  if (tid == 0) {
    // ce
    float lse1 = m1 * INV_BETA + __logf(s_f[0]);
    float ce = lse1 - row[proxy[i]] * INV_BETA;

    // part 2: assoc loss
    int   aidx[AASSOC];
    float cat[TOPN];
    const int pl = plabel[i];
    float asum = 0.0f;
    for (int j = 0; j < AASSOC; ++j) {
      aidx[j] = assoc_tab[pl * AASSOC + j];
      cat[j]  = row[aidx[j]] * INV_BETA;
      asum   += cat[j];
    }
    int nc = AASSOC, taken = 0;
    for (int t = 0; t < TOPN && taken < BG_KNN; ++t) {
      int idx = s_topIdx[t];
      bool skip = false;
      for (int j = 0; j < AASSOC; ++j) skip |= (idx == aidx[j]);
      if (!skip) { cat[nc++] = s_topVal[t] * INV_BETA; ++taken; }
    }
    float mx = cat[0];
    for (int k = 1; k < nc; ++k) mx = fmaxf(mx, cat[k]);
    float ss = 0.0f;
    for (int k = 0; k < nc; ++k) ss += __expf(cat[k] - mx);
    float assoc_loss = (mx + __logf(ss)) - asum / (float)AASSOC;

    // part 3: online loss (top-3 of per-cam tops as positives)
    unsigned long long ck[NCAMS];
    for (int c = 0; c < NCAMS; ++c) ck[c] = s_camKey[c];
    int posIdx[POS_K]; float posVal[POS_K];
    for (int r = 0; r < POS_K; ++r) {
      int bi = 0;
      for (int c = 1; c < NCAMS; ++c) if (ck[c] > ck[bi]) bi = c;
      posIdx[r] = key_idx(ck[bi]);
      posVal[r] = key_val(ck[bi]);
      ck[bi] = 0;
    }
    float cat3[POS_K + TOP_EXTRA];
    float psum = 0.0f;
    for (int r = 0; r < POS_K; ++r) { cat3[r] = posVal[r] * INV_BETA; psum += cat3[r]; }
    nc = POS_K; taken = 0;
    for (int t = 0; t < TOPN && taken < TOP_EXTRA; ++t) {
      int idx = s_topIdx[t];
      bool skip = (idx == posIdx[0]) | (idx == posIdx[1]) | (idx == posIdx[2]);
      if (!skip) { cat3[nc++] = s_topVal[t] * INV_BETA; ++taken; }
    }
    mx = cat3[0];
    for (int k = 1; k < nc; ++k) mx = fmaxf(mx, cat3[k]);
    ss = 0.0f;
    for (int k = 0; k < nc; ++k) ss += __expf(cat3[k] - mx);
    float online_loss = (mx + __logf(ss)) - psum / (float)POS_K;

    ce_out[i] = ce;
    as_out[i] = assoc_loss;
    on_out[i] = online_loss;
  }
}

// ---------------- kernel 3: per-cam mean + weighted sum ----------------
__global__ __launch_bounds__(256) void k_final(const float* __restrict__ ce,
                                               const float* __restrict__ as,
                                               const float* __restrict__ on,
                                               const int* __restrict__ cams,
                                               float* __restrict__ out) {
  __shared__ float s0[NCAMS], s1[NCAMS], s2[NCAMS], sc[NCAMS];
  const int tid = threadIdx.x;
  if (tid < NCAMS) { s0[tid] = 0.f; s1[tid] = 0.f; s2[tid] = 0.f; sc[tid] = 0.f; }
  __syncthreads();
  const int c = cams[tid];
  atomicAdd(&s0[c], ce[tid]);
  atomicAdd(&s1[c], as[tid]);
  atomicAdd(&s2[c], on[tid]);
  atomicAdd(&sc[c], 1.0f);
  __syncthreads();
  if (tid == 0) {
    float loss = 0.f;
    for (int k = 0; k < NCAMS; ++k) {
      float cnt = sc[k];
      if (cnt > 0.f)
        loss += 0.6f * (s0[k] / cnt) + 0.7f * (s1[k] / cnt) + 0.7f * (s2[k] / cnt);
    }
    out[0] = loss;
  }
}

// ---------------- launcher ----------------
extern "C" void kernel_launch(void* const* d_in, const int* in_sizes, int n_in,
                              void* d_out, int out_size, void* d_ws, size_t ws_size,
                              hipStream_t stream) {
  (void)in_sizes; (void)n_in; (void)out_size; (void)ws_size;
  const float* features = (const float*)d_in[0];
  const float* bank     = (const float*)d_in[1];
  const int*   plabel   = (const int*)d_in[2];
  const int*   proxy    = (const int*)d_in[3];
  const int*   cams     = (const int*)d_in[4];
  const int*   assoc    = (const int*)d_in[5];

  char* ws = (char*)d_ws;
  const size_t FEATBF_BYTES = (size_t)BB * DD * sizeof(unsigned short);   // 1 MB
  const size_t SIMS_BYTES   = (size_t)BB * MTOT * sizeof(float);          // 50.3 MB
  unsigned short* fb   = (unsigned short*)ws;
  float*          sims = (float*)(ws + FEATBF_BYTES);
  float*          ce   = (float*)(ws + FEATBF_BYTES + SIMS_BYTES);
  float*          as   = ce + BB;
  float*          on   = ce + 2 * BB;

  const int npairs = (BB * DD) / 2;
  k_feat_bf16<<<npairs / 256, 256, 0, stream>>>(features, (unsigned int*)fb, npairs);

  k_wmma_gemm<<<MTOT / 128, 256, 0, stream>>>(fb, bank, sims);

  k_row<<<BB, 256, 0, stream>>>(sims, proxy, cams, plabel, assoc, ce, as, on);

  k_final<<<1, 256, 0, stream>>>(ce, as, on, cams, (float*)d_out);
}
